// ChamferLossWeighted_84189948936347
// MI455X (gfx1250) — compile-verified, matrix operand order checked by reading, not run on hardware
//
#include <hip/hip_runtime.h>

// ---------------------------------------------------------------------------
// Chamfer weighted loss for MI455X (gfx1250, wave32, WMMA + TDM).
// D2(i,j) = max(|x_i|^2 + |y_j|^2 - 2 x_i.y_j, 0);  out uses sqrt(min D2).
// - fp32 split into bf16 hi/lo once (prologue); GEMM = 3 compensated
//   v_wmma_f32_16x16x32_bf16 per K-step (near-fp32 accuracy).
// - B tiles staged to LDS by the Tensor Data Mover (tensor_load_to_lds)
//   with D# padding producing a 68-dword bank-conflict-free row stride,
//   double buffered, synchronized with s_wait_tensorcnt.
// - sqrt deferred to the final reduction (min is monotone under sqrt).
// ---------------------------------------------------------------------------

typedef __attribute__((ext_vector_type(16))) __bf16       v16bf;
typedef __attribute__((ext_vector_type(8)))  float        v8f;
typedef __attribute__((ext_vector_type(4)))  unsigned int u32x4;
typedef __attribute__((ext_vector_type(8)))  int          i32x8;
typedef __attribute__((ext_vector_type(4)))  int          i32x4;

#define BS     8
#define NPTS   4096
#define CH     128
#define BROW   68            // padded LDS row stride in dwords (64 + 4 pad)
#define CTILES (NPTS / 128)

union Frag16 { v16bf v; unsigned u[8]; };

__device__ __forceinline__ unsigned short f2bf_rne(float f) {
    unsigned u = __float_as_uint(f);
    u += 0x7FFFu + ((u >> 16) & 1u);
    return (unsigned short)(u >> 16);
}

__device__ __forceinline__ void split2(float a, float b, unsigned &hi, unsigned &lo) {
    unsigned short ha = f2bf_rne(a), hb = f2bf_rne(b);
    float ra = a - __uint_as_float((unsigned)ha << 16);
    float rb = b - __uint_as_float((unsigned)hb << 16);
    hi = (unsigned)ha | ((unsigned)hb << 16);
    lo = (unsigned)f2bf_rne(ra) | ((unsigned)f2bf_rne(rb) << 16);
}

__device__ __forceinline__ unsigned lds_addr_of(const void* p) {
    // generic -> LDS(AS3) addrspacecast, then take the 32-bit LDS byte offset
    return (unsigned)(size_t)(__attribute__((address_space(3))) const void*)p;
}

// One TDM descriptor: copy a [128 rows x 64 dwords] tile (row-contiguous in
// global) into LDS, inserting 4 pad dwords after every 64 dwords -> row
// stride 68 dwords in LDS.  data_size=4B, 2D tile, no multicast.
__device__ __forceinline__ void tdm_load_tile(const unsigned* gsrc, unsigned lds_off) {
    unsigned long long ga = (unsigned long long)(size_t)gsrc;
    u32x4 g0;
    g0[0] = 1u;                                            // count=1, user mode
    g0[1] = lds_off;                                       // lds_addr[31:0]
    g0[2] = (unsigned)ga;                                  // global_addr[31:0]
    g0[3] = ((unsigned)(ga >> 32) & 0x01FFFFFFu)           // global_addr[56:32]
            | 0x80000000u;                                 // type = 2 ("image")
    i32x8 g1;
    g1[0] = (int)((2u << 16)       // data_size = 4 bytes
                | (1u << 20)       // pad_enable
                | (5u << 22)       // pad_interval: 64 dwords
                | (3u << 25));     // pad_amount:   4 dwords
    g1[1] = (int)(64u  << 16);     // tensor_dim0 = 64  (bits 79:48 low half)
    g1[2] = (int)(128u << 16);     // tensor_dim1 = 128 (bits 111:80 low half)
    g1[3] = (int)(64u  << 16);     // tile_dim0 = 64    (bits 127:112)
    g1[4] = 128;                   // tile_dim1 = 128   (bits 143:128)
    g1[5] = 64;                    // tensor_dim0_stride = 64 (bits 207:160)
    g1[6] = 0;
    g1[7] = 0;
    i32x4 z4 = {0, 0, 0, 0};
    i32x8 z8 = {0, 0, 0, 0, 0, 0, 0, 0};
    // clang-23 / therock-10.0 form: 6 args (g0, g1, g2, g3, extra, cpol)
    __builtin_amdgcn_tensor_load_to_lds(g0, g1, z4, z4, z8, 0);
}

// ---------------------------------------------------------------------------
// Kernel 1: fp32 -> packed bf16 hi/lo split for both sets + squared norms +
// colmin init.  One thread per point (128 channels).
// ---------------------------------------------------------------------------
__global__ __launch_bounds__(256) void convert_norms_kernel(
        const float* __restrict__ s1, const float* __restrict__ s2,
        unsigned* __restrict__ s1hi, unsigned* __restrict__ s1lo,
        unsigned* __restrict__ s2hi, unsigned* __restrict__ s2lo,
        float* __restrict__ x2, float* __restrict__ y2,
        unsigned* __restrict__ colminBits)
{
    const int tid  = blockIdx.x * blockDim.x + threadIdx.x;
    const int half = BS * NPTS;
    if (tid >= 2 * half) return;
    const bool isB = tid >= half;
    const int  p   = isB ? tid - half : tid;

    const float* src = (isB ? s2 : s1) + (size_t)p * CH;
    unsigned*    dh  = (isB ? s2hi : s1hi) + (size_t)p * (CH / 2);
    unsigned*    dl  = (isB ? s2lo : s1lo) + (size_t)p * (CH / 2);

    float s = 0.f;
    #pragma unroll
    for (int i = 0; i < CH / 4; ++i) {
        float4 q = ((const float4*)src)[i];
        s += q.x * q.x + q.y * q.y + q.z * q.z + q.w * q.w;
        split2(q.x, q.y, dh[2 * i],     dl[2 * i]);
        split2(q.z, q.w, dh[2 * i + 1], dl[2 * i + 1]);
    }
    if (isB) { y2[p] = s; colminBits[p] = 0x7F800000u; }
    else     { x2[p] = s; }
}

// ---------------------------------------------------------------------------
// Kernel 2: tiled WMMA distance + row/col min of clamped d^2.
// grid = (32 row-tiles, 8 batches), block = 256 threads (8 waves).
// ---------------------------------------------------------------------------
__global__ __launch_bounds__(256) void chamfer_tile_kernel(
        const unsigned* __restrict__ Ahig, const unsigned* __restrict__ Alog,
        const unsigned* __restrict__ Bhig, const unsigned* __restrict__ Blog,
        const float* __restrict__ x2, const float* __restrict__ y2,
        float* __restrict__ rowmin, unsigned* __restrict__ colminBits)
{
    __shared__ unsigned BhiL[2][128 * BROW];   // double-buffered TDM targets
    __shared__ unsigned BloL[2][128 * BROW];
    __shared__ unsigned cmin[128];

    const int b       = blockIdx.y;
    const int rowbase = blockIdx.x * 128;
    const int tid     = threadIdx.x;
    const int lane    = tid & 31;
    const int wave    = tid >> 5;
    const int l15     = lane & 15;
    const int h       = lane >> 4;     // K-half select per WMMA 16-bit layout

    const unsigned* Ah = Ahig + (size_t)b * NPTS * (CH / 2);
    const unsigned* Al = Alog + (size_t)b * NPTS * (CH / 2);
    const unsigned* Bh = Bhig + (size_t)b * NPTS * (CH / 2);
    const unsigned* Bl = Blog + (size_t)b * NPTS * (CH / 2);

    // ---- prefetch B tile 0 via TDM (wave 0 only; EXEC-independent DMA) ----
    if (tid < 32) {
        tdm_load_tile(Bh, lds_addr_of(&BhiL[0][0]));
        tdm_load_tile(Bl, lds_addr_of(&BloL[0][0]));
    }

    // ---- A fragments cached in registers for the whole block lifetime -----
    Frag16 Ahi[4], Alo[4];
    {
        const unsigned* ap = Ah + (size_t)(rowbase + wave * 16 + l15) * (CH / 2);
        const unsigned* al = Al + (size_t)(rowbase + wave * 16 + l15) * (CH / 2);
        #pragma unroll
        for (int ks = 0; ks < 4; ++ks) {
            const int idx = ks * 16 + 4 * h;       // dword index of K group 1
            *(uint4*)(&Ahi[ks].u[0]) = *(const uint4*)(ap + idx);
            *(uint4*)(&Ahi[ks].u[4]) = *(const uint4*)(ap + idx + 8);
            *(uint4*)(&Alo[ks].u[0]) = *(const uint4*)(al + idx);
            *(uint4*)(&Alo[ks].u[4]) = *(const uint4*)(al + idx + 8);
        }
    }

    float x2v[8];
    #pragma unroll
    for (int r = 0; r < 8; ++r)
        x2v[r] = x2[b * NPTS + rowbase + wave * 16 + 8 * h + r];

    float rowacc[8];                                // running min of d^2
    #pragma unroll
    for (int r = 0; r < 8; ++r) rowacc[r] = __builtin_inff();

    for (int ct = 0; ct < CTILES; ++ct) {
        __syncthreads();   // prev tile consumed: safe to refill other buffer
        if (tid < 32) {
            if (ct + 1 < CTILES) {
                const int nb = (ct + 1) & 1;
                tdm_load_tile(Bh + (size_t)(ct + 1) * 128 * (CH / 2),
                              lds_addr_of(&BhiL[nb][0]));
                tdm_load_tile(Bl + (size_t)(ct + 1) * 128 * (CH / 2),
                              lds_addr_of(&BloL[nb][0]));
                __builtin_amdgcn_s_wait_tensorcnt(2);   // tile ct complete
            } else {
                __builtin_amdgcn_s_wait_tensorcnt(0);
            }
        }
        if (tid < 128) cmin[tid] = 0x7F800000u;
        __syncthreads();   // tile ct in LDS + cmin initialized

        const unsigned* bhB = &BhiL[ct & 1][0];
        const unsigned* blB = &BloL[ct & 1][0];

        // hoist all per-tile y2 lane values: 8 loads issued up front, their
        // latency hidden behind the 96-WMMA compute body (no mid-loop waits)
        float y2v[8];
        {
            const float* y2t = y2 + b * NPTS + ct * 128 + l15;
            #pragma unroll
            for (int n = 0; n < 8; ++n) y2v[n] = y2t[n * 16];
        }

        // ---- 8 subtiles of 16x16: dot products via compensated bf16 WMMA --
        #pragma unroll
        for (int n = 0; n < 8; ++n) {
            const int pl = n * 16 + l15;           // local column point
            const unsigned* bh = bhB + pl * BROW;
            const unsigned* bl = blB + pl * BROW;

            v8f acc;
            #pragma unroll
            for (int i = 0; i < 8; ++i) acc[i] = 0.f;

            #pragma unroll
            for (int ks = 0; ks < 4; ++ks) {
                const int idx = ks * 16 + 4 * h;   // dword index of K group 1
                Frag16 bhf, blf;
                *(uint4*)(&bhf.u[0]) = *(const uint4*)(bh + idx);
                *(uint4*)(&bhf.u[4]) = *(const uint4*)(bh + idx + 8);
                *(uint4*)(&blf.u[0]) = *(const uint4*)(bl + idx);
                *(uint4*)(&blf.u[4]) = *(const uint4*)(bl + idx + 8);

                acc = __builtin_amdgcn_wmma_f32_16x16x32_bf16(
                          false, Ahi[ks].v, false, bhf.v, (short)0, acc, false, false);
                acc = __builtin_amdgcn_wmma_f32_16x16x32_bf16(
                          false, Ahi[ks].v, false, blf.v, (short)0, acc, false, false);
                acc = __builtin_amdgcn_wmma_f32_16x16x32_bf16(
                          false, Alo[ks].v, false, bhf.v, (short)0, acc, false, false);
            }

            // epilogue: clamped d^2; no sqrt here (monotone -> defer)
            float cpart = __builtin_inff();
            #pragma unroll
            for (int r = 0; r < 8; ++r) {
                float d2 = fmaxf(x2v[r] + y2v[n] - 2.0f * acc[r], 0.f);
                rowacc[r] = fminf(rowacc[r], d2);
                cpart     = fminf(cpart, d2);
            }
            cpart = fminf(cpart, __shfl_xor(cpart, 16, 32));  // merge row halves
            if (h == 0)
                atomicMin(&cmin[pl], __float_as_uint(cpart));  // d2 >= 0: bit order ok
        }
        __syncthreads();   // all waves' cmin updates done
        if (tid < 128)
            atomicMin(&colminBits[b * NPTS + ct * 128 + tid], cmin[tid]);
    }

    // ---- row-min: reduce across the 16 lanes of each half, store ----------
    #pragma unroll
    for (int mask = 1; mask <= 8; mask <<= 1) {
        #pragma unroll
        for (int r = 0; r < 8; ++r)
            rowacc[r] = fminf(rowacc[r], __shfl_xor(rowacc[r], mask, 32));
    }
    if (l15 == 0) {
        #pragma unroll
        for (int r = 0; r < 8; ++r)
            rowmin[b * NPTS + rowbase + wave * 16 + 8 * h + r] = rowacc[r];
    }
}

// ---------------------------------------------------------------------------
// Kernel 3: deterministic single-block weighted reduction (sqrt here).
// ---------------------------------------------------------------------------
__global__ __launch_bounds__(256) void final_reduce_kernel(
        const float* __restrict__ rowmin, const unsigned* __restrict__ colminBits,
        const float* __restrict__ w1, const float* __restrict__ w2,
        float* __restrict__ out)
{
    __shared__ float sdata[256];
    const int tid = threadIdx.x;
    float s = 0.f;
    for (int i = tid; i < BS * NPTS; i += 256) {
        s += w1[i] * __builtin_sqrtf(rowmin[i]);
        s += w2[i] * __builtin_sqrtf(__uint_as_float(colminBits[i]));
    }
    sdata[tid] = s;
    __syncthreads();
    for (int ofs = 128; ofs > 0; ofs >>= 1) {
        if (tid < ofs) sdata[tid] += sdata[tid + ofs];
        __syncthreads();
    }
    if (tid == 0) out[0] = sdata[0] * 0.5f;
}

// ---------------------------------------------------------------------------
extern "C" void kernel_launch(void* const* d_in, const int* in_sizes, int n_in,
                              void* d_out, int out_size, void* d_ws, size_t ws_size,
                              hipStream_t stream) {
    const float* s1 = (const float*)d_in[0];   // (8, 4096, 128) fp32
    const float* s2 = (const float*)d_in[1];   // (8, 4096, 128) fp32
    const float* w1 = (const float*)d_in[2];   // (8, 4096) fp32
    const float* w2 = (const float*)d_in[3];   // (8, 4096) fp32
    float* out = (float*)d_out;                // scalar fp32

    // workspace: s1hi | s1lo | s2hi | s2lo (each BS*NPTS*64 dwords)
    //            then x2 | y2 | rowmin | colminBits (each BS*NPTS dwords)
    const size_t big = (size_t)BS * NPTS * (CH / 2);
    unsigned* s1hi = (unsigned*)d_ws;
    unsigned* s1lo = s1hi + big;
    unsigned* s2hi = s1lo + big;
    unsigned* s2lo = s2hi + big;
    float*    x2      = (float*)(s2lo + big);
    float*    y2      = x2 + BS * NPTS;
    float*    rowmin  = y2 + BS * NPTS;
    unsigned* colBits = (unsigned*)(rowmin + BS * NPTS);

    convert_norms_kernel<<<(2 * BS * NPTS) / 256, 256, 0, stream>>>(
        s1, s2, s1hi, s1lo, s2hi, s2lo, x2, y2, colBits);

    dim3 grid(NPTS / 128, BS);
    chamfer_tile_kernel<<<grid, 256, 0, stream>>>(
        s1hi, s1lo, s2hi, s2lo, x2, y2, rowmin, colBits);

    final_reduce_kernel<<<1, 256, 0, stream>>>(rowmin, colBits, w1, w2, out);
}
